// DenseLoss_35416300323218
// MI455X (gfx1250) — matrix operand order repaired and checked by reading, not compiled, not run on hardware
//
#include <hip/hip_runtime.h>
#include <math.h>

typedef __attribute__((ext_vector_type(16))) _Float16 v16h;
typedef __attribute__((ext_vector_type(8)))  float    v8f;

#define BB 32
#define CC 64
#define NN 1024
#define INV_TEMP 10.0f
#define EPS_NORM 1e-12f
#define EPS_LOSS 1e-6f

// ---------------------------------------------------------------------------
// Kernel 1: L2-normalize along C.
//  - q_b -> f16 WMMA B-fragments (qpack), k_b -> f16 WMMA A-fragments (kpack)
//  - q_grid/k_grid -> just store reciprocal norms (dot products done on raw
//    data later; unit-dot = raw-dot * rq * rk)
// Packed layout (f16): [b][tile(64)][chunk(2)][lane(32)][elem(16)]
// ---------------------------------------------------------------------------
__global__ __launch_bounds__(256) void k_norm_pack(
    const float* __restrict__ qb, const float* __restrict__ kb,
    const float* __restrict__ qg, const float* __restrict__ kg,
    _Float16* __restrict__ qpack, _Float16* __restrict__ kpack,
    float* __restrict__ rqg, float* __restrict__ rkg)
{
    int t = blockIdx.x * 256 + threadIdx.x;     // 0 .. B*N-1
    int b = t >> 10;
    int n = t & (NN - 1);
    size_t base = (size_t)b * CC * NN + n;

    const float* pqb = qb + base;
    const float* pkb = kb + base;
    const float* pqg = qg + base;
    const float* pkg = kg + base;

    float sqb = 0.f, skb = 0.f, sqg = 0.f, skg = 0.f;
    #pragma unroll 8
    for (int c = 0; c < CC; ++c) {
        float a = pqb[c * NN]; sqb += a * a;
        float d = pkb[c * NN]; skb += d * d;
        float e = pqg[c * NN]; sqg += e * e;
        float f = pkg[c * NN]; skg += f * f;
    }
    float rnqb = 1.f / fmaxf(sqrtf(sqb), EPS_NORM);
    float rnkb = 1.f / fmaxf(sqrtf(skb), EPS_NORM);
    rqg[t] = 1.f / fmaxf(sqrtf(sqg), EPS_NORM);
    rkg[t] = 1.f / fmaxf(sqrtf(skg), EPS_NORM);

    int tile = n >> 4;
    int l    = n & 15;
    size_t fbase = ((size_t)(b * 64 + tile) * 2) * 512;  // f16 elements

    // q_b -> B-fragment: col N=l; chunk=c>>5; lane = l + 16*((c>>4)&1); elem=c&15
    #pragma unroll 8
    for (int c = 0; c < CC; ++c) {
        float v = pqb[c * NN] * rnqb;
        int lane = l + (((c >> 4) & 1) << 4);
        qpack[fbase + (size_t)(c >> 5) * 512 + lane * 16 + (c & 15)] = (_Float16)v;
    }
    // k_b -> A-fragment: row M=l; within chunk kk=c&31:
    //   lane = l + 16*((kk>>4)&1); elem = kk&15
    #pragma unroll 8
    for (int c = 0; c < CC; ++c) {
        float v = pkb[c * NN] * rnkb;
        int kk = c & 31;
        int lane = l + (((kk >> 4) & 1) << 4);
        int elem = kk & 15;
        kpack[fbase + (size_t)(c >> 5) * 512 + lane * 16 + elem] = (_Float16)v;
    }
}

// ---------------------------------------------------------------------------
// Kernel 2: fused batched GEMM (sim = k^T-rows x q-cols) + row argmax.
// One wave handles 2 i-tiles (16 rows each); loops 64 j-tiles, 2 K-chunks.
// D layout: lane<16 -> i=lane, j in {jt*16+r}; lane>=16 -> same i, j+8.
// ---------------------------------------------------------------------------
__global__ __launch_bounds__(256) void k_sim_argmax(
    const _Float16* __restrict__ qpack, const _Float16* __restrict__ kpack,
    int* __restrict__ idx)
{
    int wave = (blockIdx.x * 256 + threadIdx.x) >> 5;  // 0..1023
    int lane = threadIdx.x & 31;
    int b  = wave >> 5;                                // 32 waves per batch
    int tw = wave & 31;                                // pair of i-tiles
    int it0 = tw * 2, it1 = tw * 2 + 1;

    const v16h* qp = (const v16h*)(qpack + (size_t)b * 65536);
    const v16h* kp = (const v16h*)(kpack + (size_t)b * 65536);

    v16h b00 = qp[it0 * 64 +      lane];
    v16h b01 = qp[it0 * 64 + 32 + lane];
    v16h b10 = qp[it1 * 64 +      lane];
    v16h b11 = qp[it1 * 64 + 32 + lane];

    float best0 = -INFINITY, best1 = -INFINITY;
    int   bj0 = 0, bj1 = 0;
    int   half = (lane >= 16) ? 8 : 0;

    for (int jt = 0; jt < 64; ++jt) {
        v16h a0 = kp[jt * 64 +      lane];
        v16h a1 = kp[jt * 64 + 32 + lane];
        v8f c0 = {}; v8f c1 = {};
        c0 = __builtin_amdgcn_wmma_f32_16x16x32_f16(false, a0, false, b00,
                                                    (short)0, c0, false, false);
        c1 = __builtin_amdgcn_wmma_f32_16x16x32_f16(false, a0, false, b10,
                                                    (short)0, c1, false, false);
        c0 = __builtin_amdgcn_wmma_f32_16x16x32_f16(false, a1, false, b01,
                                                    (short)0, c0, false, false);
        c1 = __builtin_amdgcn_wmma_f32_16x16x32_f16(false, a1, false, b11,
                                                    (short)0, c1, false, false);
        int jbase = jt * 16 + half;
        #pragma unroll
        for (int r = 0; r < 8; ++r) {
            float v0 = c0[r];
            if (v0 > best0) { best0 = v0; bj0 = jbase + r; }
            float v1 = c1[r];
            if (v1 > best1) { best1 = v1; bj1 = jbase + r; }
        }
    }

    // Merge the two lane-halves (j%16 in 0..7 vs 8..15 for the same i).
    float o0 = __shfl_xor(best0, 16); int oj0 = __shfl_xor(bj0, 16);
    if (o0 > best0 || (o0 == best0 && oj0 < bj0)) { best0 = o0; bj0 = oj0; }
    float o1 = __shfl_xor(best1, 16); int oj1 = __shfl_xor(bj1, 16);
    if (o1 > best1 || (o1 == best1 && oj1 < bj1)) { best1 = o1; bj1 = oj1; }

    if (lane < 16) {
        idx[b * NN + it0 * 16 + lane] = bj0;
        idx[b * NN + it1 * 16 + lane] = bj1;
    }
}

// ---------------------------------------------------------------------------
// Kernel 3: per-row negative index (argmax over label-mismatched noise).
// ---------------------------------------------------------------------------
__global__ void k_negidx(const int* __restrict__ labels,
                         const float* __restrict__ noise,
                         int* __restrict__ negidx)
{
    int b = threadIdx.x;
    if (b >= BB) return;
    int lb = labels[b];
    float best = -INFINITY; int bi = 0;
    for (int j = 0; j < BB; ++j) {
        if (labels[j] != lb) {
            float v = noise[b * BB + j];
            if (v > best) { best = v; bi = j; }
        }
    }
    negidx[b] = bi;
}

// ---------------------------------------------------------------------------
// Kernel 4: pos/neg similarities + per-element loss, block partial sums.
// ---------------------------------------------------------------------------
__global__ __launch_bounds__(256) void k_loss(
    const float* __restrict__ qg, const float* __restrict__ kg,
    const float* __restrict__ rqg, const float* __restrict__ rkg,
    const int* __restrict__ idx, const int* __restrict__ negidx,
    float* __restrict__ partial)
{
    int t = blockIdx.x * 256 + threadIdx.x;
    int b = t >> 10;
    int i = t & (NN - 1);
    int j  = idx[t];
    int nb = negidx[b];

    const float* pq  = qg + (size_t)b  * CC * NN + i;
    const float* pkp = kg + (size_t)b  * CC * NN + j;
    const float* pkn = kg + (size_t)nb * CC * NN + i;

    float sp = 0.f, sn = 0.f;
    #pragma unroll 8
    for (int c = 0; c < CC; ++c) {
        float q = pq[c * NN];
        sp += q * pkp[c * NN];
        sn += q * pkn[c * NN];
    }
    float rq = rqg[t];
    float pos = sp * rq * rkg[b * NN + j]  * INV_TEMP;
    float neg = sn * rq * rkg[nb * NN + i] * INV_TEMP;
    float ep = expf(pos), en = expf(neg);
    float loss = -logf(ep / (ep + en + EPS_LOSS));

    __shared__ float s[256];
    s[threadIdx.x] = loss;
    __syncthreads();
    for (int off = 128; off > 0; off >>= 1) {
        if (threadIdx.x < off) s[threadIdx.x] += s[threadIdx.x + off];
        __syncthreads();
    }
    if (threadIdx.x == 0) partial[blockIdx.x] = s[0];
}

// ---------------------------------------------------------------------------
// Kernel 5: deterministic final reduction of 128 partials -> mean * LAM.
// ---------------------------------------------------------------------------
__global__ void k_final(const float* __restrict__ partial, float* __restrict__ out)
{
    __shared__ float s[128];
    s[threadIdx.x] = partial[threadIdx.x];
    __syncthreads();
    for (int off = 64; off > 0; off >>= 1) {
        if (threadIdx.x < off) s[threadIdx.x] += s[threadIdx.x + off];
        __syncthreads();
    }
    if (threadIdx.x == 0) out[0] = s[0] * (1.0f / (BB * NN));  // LAM = 1.0
}

// ---------------------------------------------------------------------------
extern "C" void kernel_launch(void* const* d_in, const int* in_sizes, int n_in,
                              void* d_out, int out_size, void* d_ws, size_t ws_size,
                              hipStream_t stream) {
    const float* qb     = (const float*)d_in[0];
    const float* kb     = (const float*)d_in[1];
    const float* qg     = (const float*)d_in[2];
    const float* kg     = (const float*)d_in[3];
    const int*   labels = (const int*)d_in[4];
    const float* noise  = (const float*)d_in[5];
    float* out = (float*)d_out;

    char* ws = (char*)d_ws;
    _Float16* qpack = (_Float16*)ws;                          // 4 MB
    _Float16* kpack = (_Float16*)(ws + (4u << 20));           // 4 MB
    float* rqg      = (float*)(ws + (8u << 20));              // 128 KB
    float* rkg      = rqg + BB * NN;                          // 128 KB
    int*   idx      = (int*)(rkg + BB * NN);                  // 128 KB
    int*   negidx   = idx + BB * NN;                          // 128 B
    float* partial  = (float*)(negidx + BB);                  // 512 B

    k_norm_pack<<<BB * NN / 256, 256, 0, stream>>>(qb, kb, qg, kg,
                                                   qpack, kpack, rqg, rkg);
    k_negidx<<<1, 32, 0, stream>>>(labels, noise, negidx);
    k_sim_argmax<<<128, 256, 0, stream>>>(qpack, kpack, idx);
    k_loss<<<BB * NN / 256, 256, 0, stream>>>(qg, kg, rqg, rkg, idx, negidx, partial);
    k_final<<<1, 128, 0, stream>>>(partial, out);
}
